// CausalSelfAttention_62440234549503
// MI455X (gfx1250) — matrix operand order repaired and checked by reading, not compiled, not run on hardware
//
#include <hip/hip_runtime.h>

typedef __bf16 bf16;
typedef __attribute__((ext_vector_type(16))) __bf16 v16bf;
typedef __attribute__((ext_vector_type(8)))  __bf16 bf16x8;
typedef __attribute__((ext_vector_type(4)))  __bf16 bf16x4;
typedef __attribute__((ext_vector_type(8)))  float  v8f;
typedef __attribute__((ext_vector_type(4)))  float  f32x4;

constexpr int Bb = 2, Ss = 2048, Dd = 2048, Hh = 16, Hd = 128;
constexpr int Mtot = Bb * Ss;          // 4096

// ---- WMMA fragment loaders (layouts per CDNA5 ISA) -------------
// A (16x32 bf16): lane half 0 -> K 0..7 & 16..23 ; lane half 1 -> K 8..15 & 24..31
__device__ __forceinline__ v16bf load_frag_a(const bf16* row, int khalf) {
  union { v16bf v; bf16x8 h[2]; } u;
  u.h[0] = *(const bf16x8*)(row + khalf);
  u.h[1] = *(const bf16x8*)(row + khalf + 16);
  return u.v;
}
// B (32x16 bf16): lane half 0 -> K 0..7 & 16..23 ; lane half 1 -> K 8..15 & 24..31
__device__ __forceinline__ v16bf load_frag_b(const bf16* col, int lane) {
  const int off = (lane >> 4) * 8;
  union { v16bf v; bf16x8 h[2]; } u;
  u.h[0] = *(const bf16x8*)(col + off);
  u.h[1] = *(const bf16x8*)(col + off + 16);
  return u.v;
}

#define WMMA_BF16(a, b, c) \
  __builtin_amdgcn_wmma_f32_16x16x32_bf16(false, (a), false, (b), (short)0, (c), false, false)

// ---- Tiled GEMM: C(MxN) = A(MxK) * Bw(KxN), bf16 WMMA, f32 accum ----------
// MODE 1: A = fp32 x, output bf16 scattered to [B,H,S,HD]
// MODE 0: A = bf16 O, output fp32 row-major (final projection)
template<int MODE>
__global__ __launch_bounds__(256)
void gemm_bf16(const void* __restrict__ Ap, const float* __restrict__ Bw,
               void* __restrict__ Outp)
{
  constexpr int BK = 32, LDT = 40;     // 40*2B = 80B rows: 16B-aligned, conflict-free
  __shared__ bf16 As[128 * LDT];
  __shared__ bf16 Bs[128 * LDT];       // stored transposed: Bs[n][k]

  const int tid   = threadIdx.x;
  const int lane  = tid & 31;
  const int wave  = tid >> 5;
  const int waveM = wave & 1;          // 2 x 64 rows
  const int waveN = wave >> 1;         // 4 x 32 cols
  const int lrow  = lane & 15;
  const int khalf = (lane >> 4) * 8;

  const int m0 = blockIdx.x * 128;
  const int n0 = blockIdx.y * 128;

  v8f acc[4][2] = {};

  for (int k0 = 0; k0 < Dd; k0 += BK) {
    // stage A tile 128x32 -> bf16 LDS (packed 8B stores)
#pragma unroll
    for (int i = 0; i < 4; ++i) {
      const int c = tid + i * 256;
      const int row = c >> 3;
      const int kc  = (c & 7) * 4;
      bf16x4 w;
      if constexpr (MODE == 1) {
        const float* A = (const float*)Ap;
        const float* ap = A + (size_t)(m0 + row) * Dd + k0 + kc;
        f32x4 v = *(const f32x4*)ap;
#pragma unroll
        for (int j = 0; j < 4; ++j) w[j] = (bf16)v[j];
        if (i == 0) __builtin_prefetch(ap + BK, 0, 3);   // next k-tile
      } else {
        const bf16* A = (const bf16*)Ap;
        const bf16* ap = A + (size_t)(m0 + row) * Dd + k0 + kc;
        w = *(const bf16x4*)ap;
        if (i == 0) __builtin_prefetch(ap + BK, 0, 3);
      }
      *(bf16x4*)(&As[row * LDT + kc]) = w;
    }
    // stage B tile 32x128 transposed -> Bs[n][k]
#pragma unroll
    for (int i = 0; i < 4; ++i) {
      const int c  = tid + i * 256;
      const int kk = c >> 5;
      const int nc = (c & 31) * 4;
      const float* bp = Bw + (size_t)(k0 + kk) * Dd + n0 + nc;
      f32x4 v = *(const f32x4*)bp;
#pragma unroll
      for (int j = 0; j < 4; ++j) Bs[(nc + j) * LDT + kk] = (bf16)v[j];
      if (i == 0) __builtin_prefetch(bp + (size_t)BK * Dd, 0, 3); // next k-tile
    }
    __syncthreads();

    v16bf af[4], bfrag[2];
#pragma unroll
    for (int mt = 0; mt < 4; ++mt)
      af[mt] = load_frag_a(&As[(waveM * 64 + mt * 16 + lrow) * LDT], khalf);
#pragma unroll
    for (int nt = 0; nt < 2; ++nt)
      bfrag[nt] = load_frag_b(&Bs[(waveN * 32 + nt * 16 + lrow) * LDT], lane);
#pragma unroll
    for (int mt = 0; mt < 4; ++mt)
#pragma unroll
      for (int nt = 0; nt < 2; ++nt)
        acc[mt][nt] = WMMA_BF16(af[mt], bfrag[nt], acc[mt][nt]);
    __syncthreads();
  }

  // epilogue: C element r of lane -> row r + (lane>>4)*8, col lane&15
#pragma unroll
  for (int mt = 0; mt < 4; ++mt)
#pragma unroll
    for (int nt = 0; nt < 2; ++nt)
#pragma unroll
      for (int r = 0; r < 8; ++r) {
        const int m = m0 + waveM * 64 + mt * 16 + r + ((lane >> 4) * 8);
        const int n = n0 + waveN * 32 + nt * 16 + lrow;
        const float val = acc[mt][nt][r];
        if constexpr (MODE == 0) {
          ((float*)Outp)[(size_t)m * Dd + n] = val;
        } else {
          const int b = m >> 11, s = m & (Ss - 1);
          const int h = n >> 7,  hd = n & (Hd - 1);
          ((bf16*)Outp)[(((size_t)b * Hh + h) * Ss + s) * Hd + hd] = (bf16)val;
        }
      }
}

// ---- RoPE (interleaved-pair variant, matches reference) -------------------
__global__ __launch_bounds__(256)
void rope_kernel(bf16* __restrict__ Qb, bf16* __restrict__ Kb)
{
  const int idx = blockIdx.x * 256 + threadIdx.x;     // over B*H*S*(HD/2)
  if (idx >= Bb * Hh * Ss * (Hd / 2)) return;
  const int i  = idx & 63;
  const int s  = (idx >> 6) & (Ss - 1);
  const int bh = idx >> 17;                           // 64*2048 = 2^17
  const size_t base = ((size_t)bh * Ss + s) * Hd;
  const int d0 = 2 * i, d1 = 2 * i + 1;
  const float L2T_64 = 0.20762050593045702f;          // log2(10000)/64
  const float f0 = exp2f(-(float)(d0 & 63) * L2T_64);
  const float f1 = exp2f(-(float)(d1 & 63) * L2T_64);
  const float sp = (float)s;
  float c0, sn0, c1, sn1;
  __sincosf(sp * f0, &sn0, &c0);
  __sincosf(sp * f1, &sn1, &c1);
  {
    const float a = (float)Qb[base + d0], b = (float)Qb[base + d1];
    Qb[base + d0] = (bf16)(a * c0 - b * sn0);
    Qb[base + d1] = (bf16)(b * c1 + a * sn1);
  }
  {
    const float a = (float)Kb[base + d0], b = (float)Kb[base + d1];
    Kb[base + d0] = (bf16)(a * c0 - b * sn0);
    Kb[base + d1] = (bf16)(b * c1 + a * sn1);
  }
}

// ---- Flash attention: 4 waves/block, 16 q-rows/wave, 32-key chunks --------
__global__ __launch_bounds__(128)
void attn_kernel(const bf16* __restrict__ Q, const bf16* __restrict__ Kg,
                 const bf16* __restrict__ V, bf16* __restrict__ O)
{
  constexpr int LDV = 40;
  __shared__ bf16 Vs[Hd * LDV];        // Vs[hd][key] (transposed for B frags)
  __shared__ bf16 Ps[4][16 * LDV];     // per-wave P tile 16x32

  const int tid   = threadIdx.x;
  const int lane  = tid & 31;
  const int wave  = tid >> 5;
  const int lrow  = lane & 15;
  const int half8 = (lane >> 4) * 8;
  const int khalf = half8;

  const int bh     = blockIdx.y;
  const int q0_blk = blockIdx.x * 64;
  const int q0     = q0_blk + wave * 16;

  const bf16* Qh = Q  + (size_t)bh * Ss * Hd;
  const bf16* Kh = Kg + (size_t)bh * Ss * Hd;
  const bf16* Vh = V  + (size_t)bh * Ss * Hd;

  // Q fragments straight from global (A layout, 128-deep K as 4 steps of 32)
  v16bf qf[4];
  {
    const bf16* qp = Qh + (size_t)(q0 + lrow) * Hd;
#pragma unroll
    for (int ks = 0; ks < 4; ++ks) qf[ks] = load_frag_a(qp + ks * 32, khalf);
  }

  float mstate[8], lstate[8];
#pragma unroll
  for (int r = 0; r < 8; ++r) { mstate[r] = -3.0e38f; lstate[r] = 0.f; }
  v8f oa[8] = {};

  const int nchunks = blockIdx.x * 2 + 2;             // keys <= q0_blk+63
  const float scale = 0.08838834764831845f;           // 1/sqrt(128)

  for (int kc = 0; kc < nchunks; ++kc) {
    const int kbase = kc * 32;
    __syncthreads();                                   // protect previous Vs
    // cooperative transpose-load of V chunk (32 keys x 128 hd)
#pragma unroll
    for (int i = 0; i < 4; ++i) {
      const int g   = tid + i * 128;
      const int kk  = g >> 4;
      const int hd8 = (g & 15) * 8;
      const bf16* vp = Vh + (size_t)(kbase + kk) * Hd + hd8;
      bf16x8 v = *(const bf16x8*)vp;
      if (i == 0) __builtin_prefetch(vp + 32 * Hd, 0, 3);   // next V chunk
#pragma unroll
      for (int j = 0; j < 8; ++j) Vs[(hd8 + j) * LDV + kk] = v[j];
    }
    __syncthreads();

    if (kbase > q0 + 15) continue;                    // wave-uniform causal skip

    // scores: S = Q (16x128) * K^T -> 16x32 (2 n-tiles)
    v8f sc[2] = {};
#pragma unroll
    for (int nt = 0; nt < 2; ++nt) {
      const bf16* kp = Kh + (size_t)(kbase + nt * 16 + lrow) * Hd;
#pragma unroll
      for (int ks = 0; ks < 4; ++ks) {
        v16bf kf = load_frag_b(kp + ks * 32, lane);
        sc[nt] = WMMA_BF16(qf[ks], kf, sc[nt]);
      }
    }

    bf16* Pw = Ps[wave];
#pragma unroll
    for (int r = 0; r < 8; ++r) {
      const int qrow = q0 + r + half8;
      float s0 = sc[0][r] * scale;
      float s1 = sc[1][r] * scale;
      if (kbase + lrow      > qrow) s0 = -INFINITY;
      if (kbase + 16 + lrow > qrow) s1 = -INFINITY;
      float mx = fmaxf(s0, s1);
#pragma unroll
      for (int off = 1; off < 16; off <<= 1)
        mx = fmaxf(mx, __shfl_xor(mx, off, 32));
      const float mo  = mstate[r];
      const float mn  = fmaxf(mo, mx);
      const float mns = (mn > -1.0e37f) ? mn : 0.0f;
      const float corr = __expf(mo - mns);
      const float p0 = __expf(s0 - mns);
      const float p1 = __expf(s1 - mns);
      mstate[r] = mn;
      float rs = p0 + p1;
#pragma unroll
      for (int off = 1; off < 16; off <<= 1)
        rs += __shfl_xor(rs, off, 32);
      lstate[r] = lstate[r] * corr + rs;
#pragma unroll
      for (int t = 0; t < 8; ++t) oa[t][r] *= corr;
      const int prow = r + half8;
      Pw[prow * LDV + lrow]      = (bf16)p0;
      Pw[prow * LDV + 16 + lrow] = (bf16)p1;
    }

    // O += P (16x32) * V (32x128): 8 hd-tiles
    v16bf pf = load_frag_a(&Pw[lrow * LDV], khalf);
#pragma unroll
    for (int t = 0; t < 8; ++t) {
      v16bf vf = load_frag_b(&Vs[(t * 16 + lrow) * LDV], lane);
      oa[t] = WMMA_BF16(pf, vf, oa[t]);
    }
  }

  // epilogue: O[b, s, h*HD + hd] bf16, row-major (B*S) x D for final GEMM
  const int b = bh >> 4;
  const int h = bh & (Hh - 1);
  bf16* Ob = O + (size_t)b * Ss * Dd + (size_t)h * Hd;
#pragma unroll
  for (int r = 0; r < 8; ++r) {
    const int srow = q0 + r + half8;
    const float inv = 1.0f / lstate[r];
#pragma unroll
    for (int t = 0; t < 8; ++t) {
      const int hd = t * 16 + lrow;
      Ob[(size_t)srow * Dd + hd] = (bf16)(oa[t][r] * inv);
    }
  }
}

// ---- host orchestration ---------------------------------------------------
extern "C" void kernel_launch(void* const* d_in, const int* in_sizes, int n_in,
                              void* d_out, int out_size, void* d_ws, size_t ws_size,
                              hipStream_t stream)
{
  (void)in_sizes; (void)n_in; (void)out_size; (void)ws_size;
  const float* x  = (const float*)d_in[0];
  const float* Wq = (const float*)d_in[1];
  const float* Wk = (const float*)d_in[2];
  const float* Wv = (const float*)d_in[3];
  const float* Wo = (const float*)d_in[4];

  const size_t elems = (size_t)Bb * Ss * Dd;   // 8M elems (16 MB bf16 each)
  bf16* Qb = (bf16*)d_ws;
  bf16* Kb = Qb + elems;
  bf16* Vb = Kb + elems;
  bf16* Ob = Vb + elems;

  dim3 gg(Mtot / 128, Dd / 128);               // 32 x 16 blocks
  gemm_bf16<1><<<gg, 256, 0, stream>>>((const void*)x, Wq, (void*)Qb);
  gemm_bf16<1><<<gg, 256, 0, stream>>>((const void*)x, Wk, (void*)Kb);
  gemm_bf16<1><<<gg, 256, 0, stream>>>((const void*)x, Wv, (void*)Vb);

  rope_kernel<<<(Bb * Hh * Ss * (Hd / 2)) / 256, 256, 0, stream>>>(Qb, Kb);

  attn_kernel<<<dim3(Ss / 64, Bb * Hh), 128, 0, stream>>>(Qb, Kb, Vb, Ob);

  gemm_bf16<0><<<gg, 256, 0, stream>>>((const void*)Ob, Wo, d_out);
}
